// VAE_decoder_77240691851373
// MI455X (gfx1250) — compile-verified, matrix operand order checked by reading, not run on hardware
//
#include <hip/hip_runtime.h>
#include <hip/hip_bf16.h>
#include <stdint.h>

// ---------------------------------------------------------------------------
// MoE VAE decoder for MI455X (gfx1250, wave32, WMMA).
// GEMMs: LDS double-buffered via GLOBAL_LOAD_ASYNC_TO_LDS_B128 (ASYNCcnt),
// compute via v_wmma_f32_16x16x32_f16 with f32 accumulation.
// Single barrier per K-step; precomputed staging addresses.
// ---------------------------------------------------------------------------

typedef __attribute__((ext_vector_type(16))) _Float16 v16h;
typedef __attribute__((ext_vector_type(8)))  _Float16 v8h;
typedef __attribute__((ext_vector_type(8)))  float    v8f;

#define B_    16384
#define MOT_  128
#define LAT_  32
#define IN_   288      // MOTION*(USED-1) + LAT = 256 + 32
#define H1_   512
#define H2_   512
#define MOE_  8
#define K2_   (H1_ + LAT_)   // 544
#define K3_   (H2_ + LAT_)   // 544

__device__ __forceinline__ float elu_f(float v) {
    return v > 0.0f ? v : (__expf(v) - 1.0f);
}

// Low 32 bits of a generic pointer to __shared__ == LDS byte address
__device__ __forceinline__ uint32_t lds_addr32(const void* p) {
    return (uint32_t)(uintptr_t)p;
}

// Issue one 16-byte async global->LDS copy (tracked with ASYNCcnt).
__device__ __forceinline__ void async_copy_b128(uint32_t lds_addr, const void* gptr) {
    asm volatile("global_load_async_to_lds_b128 %0, %1, off"
                 :: "v"(lds_addr), "v"((uint64_t)(uintptr_t)gptr)
                 : "memory");
}

__device__ __forceinline__ void wait_async0() {
    asm volatile("s_wait_asynccnt 0x0" ::: "memory");
}

// ---------------------------------------------------------------------------
// Pack x = concat(motions, z) -> f16 [B, 288]
// ---------------------------------------------------------------------------
__global__ void pack_x_kernel(const float* __restrict__ motions,
                              const float* __restrict__ z,
                              _Float16* __restrict__ xh) {
    int idx = blockIdx.x * blockDim.x + threadIdx.x;
    if (idx >= B_ * IN_) return;
    int b = idx / IN_;
    int c = idx - b * IN_;
    float v = (c < 256) ? motions[(size_t)b * 256 + c] : z[(size_t)b * LAT_ + (c - 256)];
    xh[idx] = (_Float16)v;
}

// Write z (f16) into the tail columns [512..543] of h1cat and h2cat [B,544]
__global__ void pack_ztail_kernel(const float* __restrict__ z,
                                  _Float16* __restrict__ h1cat,
                                  _Float16* __restrict__ h2cat) {
    int idx = blockIdx.x * blockDim.x + threadIdx.x;
    if (idx >= B_ * LAT_) return;
    int b = idx / LAT_;
    int j = idx - b * LAT_;
    _Float16 v = (_Float16)z[idx];
    h1cat[(size_t)b * K2_ + H1_ + j] = v;
    h2cat[(size_t)b * K3_ + H2_ + j] = v;
}

// ---------------------------------------------------------------------------
// Convert f32 weight [E][K][N] -> f16 transposed [E][Npad][K] (pad rows = 0)
// ---------------------------------------------------------------------------
__global__ void convtrans_kernel(const float* __restrict__ src,
                                 _Float16* __restrict__ dst,
                                 int K, int N, int Npad) {
    int n = blockIdx.x * 16 + threadIdx.x;
    int k = blockIdx.y * 16 + threadIdx.y;
    int e = blockIdx.z;
    if (k >= K || n >= Npad) return;
    float v = (n < N) ? src[(size_t)e * K * N + (size_t)k * N + n] : 0.0f;
    dst[(size_t)e * Npad * K + (size_t)n * K + k] = (_Float16)v;
}

// ---------------------------------------------------------------------------
// WMMA GEMM with LDS double-buffered async staging.
//   C[M,N] = act(A[M,K] * Bt[N,K]^T + bias[N])
//   A  : f16 row-major, leading dim lda
//   Bt : f16 "N-major" (transposed weight), leading dim ldb == K
// EPI = 0 : ELU -> f16 store (outH, ldcH)                [hidden layers]
// EPI = 1 : ELU -> f32 store, cols < Nreal only (outF)   [gate layer 3]
// EPI = 2 : ELU * para[row,expert] accumulated into outF [expert layer 3]
// ---------------------------------------------------------------------------
template<int WAVES_M, int WAVES_N, int WMT, int WNT, int EPI>
__global__ __launch_bounds__(WAVES_M * WAVES_N * 32)
void gemm_wmma_kernel(const _Float16* __restrict__ A, int lda,
                      const _Float16* __restrict__ Bt, int ldb,
                      const float* __restrict__ bias,
                      int K, int Nreal,
                      _Float16* __restrict__ outH, int ldcH,
                      float* __restrict__ outF, int ldcF,
                      const float* __restrict__ para, int expert, int initAcc) {
    constexpr int BM = WAVES_M * WMT * 16;
    constexpr int BN = WAVES_N * WNT * 16;
    constexpr int NTHREADS = WAVES_M * WAVES_N * 32;
    constexpr int A_CH = BM * 4;                      // 16B chunks per stage
    constexpr int B_CH = BN * 4;
    constexpr int NCA = (A_CH + NTHREADS - 1) / NTHREADS;
    constexpr int NCB = (B_CH + NTHREADS - 1) / NTHREADS;

    __shared__ _Float16 sA[2][BM * 32];   // [stage][row*32 + k]
    __shared__ _Float16 sB[2][BN * 32];   // [stage][col*32 + k]

    const int tid   = threadIdx.x;
    const int lane  = tid & 31;
    const int wave  = tid >> 5;
    const int waveM = wave % WAVES_M;
    const int waveN = wave / WAVES_M;

    const int blockM = blockIdx.y * BM;
    const int blockN = blockIdx.x * BN;
    const int mBase  = blockM + waveM * WMT * 16;
    const int nBase  = blockN + waveN * WNT * 16;

    const int mrow = lane & 15;   // row (A) / col (B) within 16-wide tile
    const int csel = lane >> 4;   // K-chunk select per the 16-bit A layout

    union F16x16 { v16h v; v8h h[2]; };

    v8f acc[WMT][WNT];
    #pragma unroll
    for (int i = 0; i < WMT; ++i)
        #pragma unroll
        for (int j = 0; j < WNT; ++j)
            acc[i][j] = (v8f){};

    // ---- precomputed staging addresses (advance global ptrs per stage) ----
    const _Float16* aGp[NCA];
    uint32_t        aL0[NCA];
    #pragma unroll
    for (int u = 0; u < NCA; ++u) {
        const int c = tid + u * NTHREADS;
        const int row = c >> 2, seg = c & 3;          // seg = 8-elem chunk
        aGp[u] = A + (size_t)(blockM + row) * lda + seg * 8;
        aL0[u] = lds_addr32(&sA[0][row * 32 + seg * 8]);
    }
    const _Float16* bGp[NCB];
    uint32_t        bL0[NCB];
    #pragma unroll
    for (int u = 0; u < NCB; ++u) {
        const int c = tid + u * NTHREADS;
        const int row = (c >> 2) % BN, seg = c & 3;
        bGp[u] = Bt + (size_t)(blockN + row) * ldb + seg * 8;
        bL0[u] = lds_addr32(&sB[0][row * 32 + seg * 8]);
    }

    auto stageLoad = [&](int st, bool pf) {
        const uint32_t aOff = st ? (uint32_t)(BM * 64) : 0u;   // bytes
        const uint32_t bOff = st ? (uint32_t)(BN * 64) : 0u;
        #pragma unroll
        for (int u = 0; u < NCA; ++u) {
            async_copy_b128(aL0[u] + aOff, aGp[u]);
            aGp[u] += 32;
        }
        #pragma unroll
        for (int u = 0; u < NCB; ++u) {
            if (NTHREADS * (u + 1) <= B_CH || tid < B_CH - u * NTHREADS) {
                async_copy_b128(bL0[u] + bOff, bGp[u]);
                bGp[u] += 32;
            }
        }
        if (pf) __builtin_prefetch(aGp[0] + 32, 0, 0);   // one K-step ahead
    };

    const int nk = K >> 5;   // K is always a multiple of 32 here
    stageLoad(0, nk > 1);

    for (int ks = 0; ks < nk; ++ks) {
        const int st = ks & 1;
        wait_async0();        // this wave's staged copies have landed in LDS
        __syncthreads();      // all copies landed; all readers of 1-st done
        if (ks + 1 < nk) stageLoad(1 - st, ks + 2 < nk);

        // Fragment loads from LDS (ds_load_b128) + WMMA
        F16x16 af[WMT], bf[WNT];
        const _Float16* aBase = &sA[st][(waveM * WMT * 16 + mrow) * 32 + csel * 8];
        const _Float16* bBase = &sB[st][(waveN * WNT * 16 + mrow) * 32 + csel * 8];
        #pragma unroll
        for (int i = 0; i < WMT; ++i) {
            af[i].h[0] = *(const v8h*)(aBase + i * 16 * 32);
            af[i].h[1] = *(const v8h*)(aBase + i * 16 * 32 + 16);
        }
        #pragma unroll
        for (int j = 0; j < WNT; ++j) {
            bf[j].h[0] = *(const v8h*)(bBase + j * 16 * 32);
            bf[j].h[1] = *(const v8h*)(bBase + j * 16 * 32 + 16);
        }
        #pragma unroll
        for (int i = 0; i < WMT; ++i)
            #pragma unroll
            for (int j = 0; j < WNT; ++j)
                acc[i][j] = __builtin_amdgcn_wmma_f32_16x16x32_f16(
                    /*neg_a=*/false, af[i].v, /*neg_b=*/false, bf[j].v,
                    /*c_mod=*/(short)0, acc[i][j],
                    /*reuse_a=*/false, /*reuse_b=*/false);
        // No trailing barrier: a wave only passes the next top barrier after
        // its ds reads of buffer st completed (s_wait_dscnt before the WMMAs),
        // so overwriting st after that barrier is safe.
    }

    // Epilogue. C layout: lanes 0-15 -> M = vgpr index, lanes 16-31 -> M = 8+idx.
    #pragma unroll
    for (int i = 0; i < WMT; ++i) {
        const int rowTop = mBase + i * 16 + csel * 8;
        #pragma unroll
        for (int j = 0; j < WNT; ++j) {
            const int col = nBase + j * 16 + (lane & 15);
            float bcol;
            if (EPI == 1) bcol = (col < Nreal) ? bias[col] : 0.0f;
            else          bcol = bias[col];
            #pragma unroll
            for (int r = 0; r < 8; ++r) {
                const int row = rowTop + r;
                float v = elu_f(acc[i][j][r] + bcol);
                if (EPI == 0) {
                    outH[(size_t)row * ldcH + col] = (_Float16)v;
                } else if (EPI == 1) {
                    if (col < Nreal) outF[(size_t)row * ldcF + col] = v;
                } else {
                    const float  p   = para[(size_t)row * MOE_ + expert];
                    const size_t off = (size_t)row * ldcF + col;
                    const float  prv = initAcc ? 0.0f : outF[off];
                    outF[off] = prv + p * v;
                }
            }
        }
    }
}

// ---------------------------------------------------------------------------
// Frobenius-norm^2 reduction of para [B, MOE]
// ---------------------------------------------------------------------------
__global__ void zero_scalar_kernel(float* p) { *p = 0.0f; }

__global__ void reduce_sq_kernel(const float* __restrict__ para,
                                 float* __restrict__ out, int n) {
    __shared__ float s[256];
    float v = 0.0f;
    for (int i = blockIdx.x * blockDim.x + threadIdx.x; i < n;
         i += gridDim.x * blockDim.x) {
        float p = para[i];
        v += p * p;
    }
    s[threadIdx.x] = v;
    __syncthreads();
    for (int off = 128; off > 0; off >>= 1) {
        if (threadIdx.x < off) s[threadIdx.x] += s[threadIdx.x + off];
        __syncthreads();
    }
    if (threadIdx.x == 0) atomicAdd(out, s[0]);
}

// out = sigmoid(acc / sqrt(normsq))
__global__ void finalize_kernel(const float* __restrict__ acc,
                                const float* __restrict__ normsq,
                                float* __restrict__ out) {
    int idx = blockIdx.x * blockDim.x + threadIdx.x;
    if (idx >= B_ * MOT_) return;
    float inv = rsqrtf(normsq[0]);
    float v = acc[idx] * inv;
    out[idx] = 1.0f / (1.0f + __expf(-v));
}

// ---------------------------------------------------------------------------
// Host launch
// ---------------------------------------------------------------------------
extern "C" void kernel_launch(void* const* d_in, const int* in_sizes, int n_in,
                              void* d_out, int out_size, void* d_ws, size_t ws_size,
                              hipStream_t stream) {
    const float* motions = (const float*)d_in[0];
    const float* z       = (const float*)d_in[1];
    const float* gw1 = (const float*)d_in[2];
    const float* gb1 = (const float*)d_in[3];
    const float* gw2 = (const float*)d_in[4];
    const float* gb2 = (const float*)d_in[5];
    const float* gw3 = (const float*)d_in[6];
    const float* gb3 = (const float*)d_in[7];
    const float* W1  = (const float*)d_in[8];
    const float* b1  = (const float*)d_in[9];
    const float* W2  = (const float*)d_in[10];
    const float* b2  = (const float*)d_in[11];
    const float* W3  = (const float*)d_in[12];
    const float* b3  = (const float*)d_in[13];
    float* out = (float*)d_out;

    // ---- workspace carve-up (256B aligned) ----
    char* wsb = (char*)d_ws;
    size_t o = 0;
    auto alloc = [&](size_t bytes) {
        size_t r = o;
        o = (o + bytes + 255) & ~(size_t)255;
        return r;
    };
    const size_t off_xh   = alloc((size_t)B_ * IN_  * 2);  // x     f16 [B,288]
    const size_t off_g1   = alloc((size_t)B_ * H1_  * 2);  // gate1 f16 [B,512]
    const size_t off_g2   = alloc((size_t)B_ * H2_  * 2);  // gate2 f16 [B,512]
    const size_t off_para = alloc((size_t)B_ * MOE_ * 4);  // para  f32 [B,8]
    const size_t off_norm = alloc(4);                       // ||para||_F^2
    const size_t off_h1   = alloc((size_t)B_ * K2_ * 2);   // [h1|z] f16 [B,544]
    const size_t off_h2   = alloc((size_t)B_ * K3_ * 2);   // [h2|z] f16 [B,544]
    const size_t off_acc  = alloc((size_t)B_ * MOT_ * 4);  // sum_e para*h3 f32
    const size_t off_gw1t = alloc((size_t)H1_ * IN_ * 2);  // [512,288]
    const size_t off_gw2t = alloc((size_t)H2_ * H1_ * 2);  // [512,512]
    const size_t off_gw3t = alloc((size_t)16 * H2_ * 2);   // [16,512] padded
    const size_t off_W1t  = alloc((size_t)MOE_ * H1_ * IN_ * 2);  // [8][512,288]
    const size_t off_W2t  = alloc((size_t)MOE_ * H2_ * K2_ * 2);  // [8][512,544]
    const size_t off_W3t  = alloc((size_t)MOE_ * MOT_ * K3_ * 2); // [8][128,544]
    (void)ws_size; (void)in_sizes; (void)n_in; (void)out_size;

    _Float16* xh    = (_Float16*)(wsb + off_xh);
    _Float16* g1    = (_Float16*)(wsb + off_g1);
    _Float16* g2    = (_Float16*)(wsb + off_g2);
    float*    para  = (float*)(wsb + off_para);
    float*    norm  = (float*)(wsb + off_norm);
    _Float16* h1    = (_Float16*)(wsb + off_h1);
    _Float16* h2    = (_Float16*)(wsb + off_h2);
    float*    accb  = (float*)(wsb + off_acc);
    _Float16* gw1t  = (_Float16*)(wsb + off_gw1t);
    _Float16* gw2t  = (_Float16*)(wsb + off_gw2t);
    _Float16* gw3t  = (_Float16*)(wsb + off_gw3t);
    _Float16* W1t   = (_Float16*)(wsb + off_W1t);
    _Float16* W2t   = (_Float16*)(wsb + off_W2t);
    _Float16* W3t   = (_Float16*)(wsb + off_W3t);

    // ---- pack activations ----
    {
        int n = B_ * IN_;
        pack_x_kernel<<<(n + 255) / 256, 256, 0, stream>>>(motions, z, xh);
        int m = B_ * LAT_;
        pack_ztail_kernel<<<(m + 255) / 256, 256, 0, stream>>>(z, h1, h2);
    }

    // ---- convert + transpose weights to f16 [N,K] ----
    {
        dim3 blk(16, 16, 1);
        convtrans_kernel<<<dim3((H1_ + 15) / 16, (IN_ + 15) / 16, 1), blk, 0, stream>>>(gw1, gw1t, IN_, H1_, H1_);
        convtrans_kernel<<<dim3((H2_ + 15) / 16, (H1_ + 15) / 16, 1), blk, 0, stream>>>(gw2, gw2t, H1_, H2_, H2_);
        convtrans_kernel<<<dim3(1,              (H2_ + 15) / 16, 1), blk, 0, stream>>>(gw3, gw3t, H2_, MOE_, 16);
        convtrans_kernel<<<dim3((H1_ + 15) / 16, (IN_ + 15) / 16, MOE_), blk, 0, stream>>>(W1, W1t, IN_, H1_, H1_);
        convtrans_kernel<<<dim3((H2_ + 15) / 16, (K2_ + 15) / 16, MOE_), blk, 0, stream>>>(W2, W2t, K2_, H2_, H2_);
        convtrans_kernel<<<dim3((MOT_ + 15) / 16, (K3_ + 15) / 16, MOE_), blk, 0, stream>>>(W3, W3t, K3_, MOT_, MOT_);
    }

    // ---- gate MLP ----
    {
        dim3 blk(256, 1, 1);
        dim3 grd512(H1_ / 128, B_ / 128, 1);
        gemm_wmma_kernel<4, 2, 2, 4, 0><<<grd512, blk, 0, stream>>>(
            xh, IN_, gw1t, IN_, gb1, IN_, H1_, g1, H1_, nullptr, 0, nullptr, 0, 0);
        gemm_wmma_kernel<4, 2, 2, 4, 0><<<grd512, blk, 0, stream>>>(
            g1, H1_, gw2t, H1_, gb2, H1_, H2_, g2, H2_, nullptr, 0, nullptr, 0, 0);
        dim3 grd16(1, B_ / 128, 1);
        gemm_wmma_kernel<8, 1, 1, 1, 1><<<grd16, blk, 0, stream>>>(
            g2, H2_, gw3t, H2_, gb3, H2_, MOE_, nullptr, 0, para, MOE_, nullptr, 0, 0);
    }

    // ---- ||para||_F^2 ----
    zero_scalar_kernel<<<1, 1, 0, stream>>>(norm);
    reduce_sq_kernel<<<256, 256, 0, stream>>>(para, norm, B_ * MOE_);

    // ---- experts (sequential, accumulate para-weighted outputs) ----
    {
        dim3 blk(256, 1, 1);
        dim3 grd512(H1_ / 128, B_ / 128, 1);
        dim3 grd128(MOT_ / 128, B_ / 128, 1);
        for (int e = 0; e < MOE_; ++e) {
            const _Float16* w1e = W1t + (size_t)e * H1_ * IN_;
            const _Float16* w2e = W2t + (size_t)e * H2_ * K2_;
            const _Float16* w3e = W3t + (size_t)e * MOT_ * K3_;
            gemm_wmma_kernel<4, 2, 2, 4, 0><<<grd512, blk, 0, stream>>>(
                xh, IN_, w1e, IN_, b1 + (size_t)e * H1_, IN_, H1_,
                h1, K2_, nullptr, 0, nullptr, 0, 0);
            gemm_wmma_kernel<4, 2, 2, 4, 0><<<grd512, blk, 0, stream>>>(
                h1, K2_, w2e, K2_, b2 + (size_t)e * H2_, K2_, H2_,
                h2, K3_, nullptr, 0, nullptr, 0, 0);
            gemm_wmma_kernel<4, 2, 2, 4, 2><<<grd128, blk, 0, stream>>>(
                h2, K3_, w3e, K3_, b3 + (size_t)e * MOT_, K3_, MOT_,
                nullptr, 0, accb, MOT_, para, e, (e == 0) ? 1 : 0);
        }
    }

    // ---- normalize + sigmoid ----
    {
        int n = B_ * MOT_;
        finalize_kernel<<<(n + 255) / 256, 256, 0, stream>>>(accb, norm, out);
    }
}
